// GAT_1503238553771
// MI455X (gfx1250) — compile-verified, matrix operand order checked by reading, not run on hardware
//
#include <hip/hip_runtime.h>

#define NN 50000
#define NE 400000

typedef __attribute__((ext_vector_type(16))) __bf16 v16bf;
typedef __attribute__((ext_vector_type(8)))  __bf16 v8bf;
typedef __attribute__((ext_vector_type(8)))  float  v8f;

// ---------------------------------------------------------------------------
// Fill a buffer with a 32-bit pattern (0 for zeros, 0xFF800000 for -inf).
// ---------------------------------------------------------------------------
__global__ void fill_u32(unsigned int* __restrict__ p, unsigned int v, int n) {
    int i = blockIdx.x * blockDim.x + threadIdx.x;
    if (i < n) p[i] = v;
}

// ---------------------------------------------------------------------------
// Convert an fp32 panel to bf16 row-major (A operand for the GEMMs).
// ---------------------------------------------------------------------------
__global__ void pack_A_bf16(const float* __restrict__ A, __bf16* __restrict__ Ap,
                            int total) {
    int i = blockIdx.x * blockDim.x + threadIdx.x;
    if (i < total) Ap[i] = (__bf16)A[i];
}

// ---------------------------------------------------------------------------
// Pre-pack W[K,N] fp32 into the per-lane WMMA B-fragment layout, bf16:
//   Wp[(kt*(N/16)+nt)*512 + lane*16 + j]
// lane holds column nt*16+(lane&15); j0..7 -> K=kt*32+kg+j, j8..15 -> +16.
// One thread per (tile, lane); writes 16 contiguous bf16 (32B).
// ---------------------------------------------------------------------------
__global__ void pack_W_bf16(const float* __restrict__ W, __bf16* __restrict__ Wp,
                            int K, int N) {
    int t = blockIdx.x * blockDim.x + threadIdx.x;
    int total = (K / 32) * (N / 16) * 32;
    if (t >= total) return;
    int lane = t & 31;
    int tile = t >> 5;
    int nt   = tile % (N / 16);
    int kt   = tile / (N / 16);
    int col  = nt * 16 + (lane & 15);
    int kg   = (lane >> 4) * 8;
    int kb   = kt * 32 + kg;
    __bf16* out = Wp + (size_t)tile * 512 + lane * 16;
#pragma unroll
    for (int j = 0; j < 8; ++j) {
        out[j]     = (__bf16)W[(size_t)(kb + j) * N + col];
        out[8 + j] = (__bf16)W[(size_t)(kb + 16 + j) * N + col];
    }
}

// ---------------------------------------------------------------------------
// C[M,N] = Ap[M,K](bf16) @ Wp(packed bf16), f32 accumulate via WMMA.
// Block = 128 threads = 4 waves; wave w owns rows [blockIdx.x*64+16w, +16)
// and a 16x64 strip of columns (4 accumulators, A fragment reused 4x).
// Requires K%32==0, N%64==0; M%16==0 with wave-uniform tail guard.
// ---------------------------------------------------------------------------
__global__ __launch_bounds__(128) void gemm_wmma_bf16p(
    const __bf16* __restrict__ Ap, const __bf16* __restrict__ Wp,
    float* __restrict__ C, int M, int N, int K)
{
    const int lane  = threadIdx.x & 31;
    const int wave  = threadIdx.x >> 5;
    const int mTile = blockIdx.x * 64 + wave * 16;
    if (mTile >= M) return;               // wave-uniform: EXEC stays all-1s
    const int nBase = blockIdx.y * 64;
    const int r16   = lane & 15;
    const int kg    = (lane >> 4) * 8;
    const int ntTot = N >> 4;

    v8f acc0 = {}, acc1 = {}, acc2 = {}, acc3 = {};
    for (int k0 = 0; k0 < K; k0 += 32) {
        const v8bf* ap = (const v8bf*)(Ap + (size_t)(mTile + r16) * K + k0 + kg);
        v8bf alo = ap[0], ahi = ap[2];    // two contiguous b128 loads
        v16bf a = __builtin_shufflevector(alo, ahi,
                      0,1,2,3,4,5,6,7,8,9,10,11,12,13,14,15);
        const __bf16* wrow = Wp +
            ((size_t)(k0 >> 5) * ntTot + (nBase >> 4)) * 512 + lane * 16;
#pragma unroll
        for (int nt = 0; nt < 4; ++nt) {
            const v8bf* bp = (const v8bf*)(wrow + (size_t)nt * 512);
            v8bf blo = bp[0], bhi = bp[1];
            v16bf b = __builtin_shufflevector(blo, bhi,
                          0,1,2,3,4,5,6,7,8,9,10,11,12,13,14,15);
            v8f c = (nt == 0) ? acc0 : (nt == 1) ? acc1 : (nt == 2) ? acc2 : acc3;
            c = __builtin_amdgcn_wmma_f32_16x16x32_bf16(
                    false, a, false, b, (short)0, c, false, false);
            if      (nt == 0) acc0 = c;
            else if (nt == 1) acc1 = c;
            else if (nt == 2) acc2 = c;
            else              acc3 = c;
        }
    }
    const int rowBase = mTile + (lane >> 4) * 8;
#pragma unroll
    for (int nt = 0; nt < 4; ++nt) {
        const v8f c = (nt == 0) ? acc0 : (nt == 1) ? acc1 : (nt == 2) ? acc2 : acc3;
        const int col = nBase + nt * 16 + r16;
#pragma unroll
        for (int r = 0; r < 8; ++r)
            C[(size_t)(rowBase + r) * N + col] = c[r];
    }
}

// ---------------------------------------------------------------------------
// el[n,h] = sum_d f[n,h,d]*al[h,d];  er likewise.  One thread per (n,h).
// ---------------------------------------------------------------------------
__global__ void attn_logits(const float* __restrict__ f,
                            const float* __restrict__ al,
                            const float* __restrict__ ar,
                            float* __restrict__ el, float* __restrict__ er,
                            int NH, int H, int D)
{
    int idx = blockIdx.x * blockDim.x + threadIdx.x;
    if (idx >= NH) return;
    int h = idx % H;
    const float4* fp = (const float4*)(f + (size_t)idx * D);
    const float4* ap = (const float4*)(al + (size_t)h * D);
    const float4* bp = (const float4*)(ar + (size_t)h * D);
    float sl = 0.f, sr = 0.f;
    for (int j = 0; j < D / 4; ++j) {
        float4 fv = fp[j], av = ap[j], bv = bp[j];
        sl += fv.x*av.x + fv.y*av.y + fv.z*av.z + fv.w*av.w;
        sr += fv.x*bv.x + fv.y*bv.y + fv.z*bv.z + fv.w*bv.w;
    }
    el[idx] = sl; er[idx] = sr;
}

// Exact float atomic max via signed/unsigned int punning (init = -inf bits).
__device__ inline void atomicMaxF(float* addr, float val) {
    if (val >= 0.f) atomicMax((int*)addr, __float_as_int(val));
    else            atomicMin((unsigned int*)addr,
                              (unsigned int)__float_as_int(val));
}

// ---------------------------------------------------------------------------
// Pass 1: e = leaky_relu(el[src]+er[dst]); segment-max into mbuf[dst,h].
// ---------------------------------------------------------------------------
__global__ void edge_logits_max(float* __restrict__ ebuf,
                                const float* __restrict__ el,
                                const float* __restrict__ er,
                                const int* __restrict__ src,
                                const int* __restrict__ dst,
                                float* __restrict__ mbuf, int EH, int H)
{
    int idx = blockIdx.x * blockDim.x + threadIdx.x;
    if (idx >= EH) return;
    int e = idx / H, h = idx - e * H;
    int s = src[e], d = dst[e];
    float v = el[s * H + h] + er[d * H + h];
    v = v > 0.f ? v : 0.2f * v;           // leaky relu, slope 0.2
    ebuf[idx] = v;
    atomicMaxF(mbuf + d * H + h, v);
}

// ---------------------------------------------------------------------------
// Pass 2: ex = exp(e - m[dst]); segment-sum into denom[dst,h]; store ex.
// ---------------------------------------------------------------------------
__global__ void edge_exp_sum(float* __restrict__ ebuf,
                             const int* __restrict__ dst,
                             const float* __restrict__ mbuf,
                             float* __restrict__ denom, int EH, int H)
{
    int idx = blockIdx.x * blockDim.x + threadIdx.x;
    if (idx >= EH) return;
    int e = idx / H, h = idx - e * H;
    int d = dst[e];
    float ex = __expf(ebuf[idx] - mbuf[d * H + h]);
    ebuf[idx] = ex;
    atomicAdd(denom + d * H + h, ex);
}

// ---------------------------------------------------------------------------
// Pass 3: rst[dst,h,:] += alpha * f[src,h,:].  Flattened over E*(H*D/4):
// each thread does one b128 gather + 4 global_atomic_add_f32 (L2-resident).
// D=64 so a float4 never crosses a head boundary.
// ---------------------------------------------------------------------------
__global__ void edge_aggregate(float* __restrict__ rst,
                               const float* __restrict__ f,
                               const float* __restrict__ ebuf,
                               const float* __restrict__ denom,
                               const int* __restrict__ src,
                               const int* __restrict__ dst,
                               int H, int D, int E)
{
    int q = (H * D) >> 2;                 // float4 slots per edge
    int t = blockIdx.x * blockDim.x + threadIdx.x;
    int e = t / q;
    if (e >= E) return;
    int s4 = t - e * q;
    int hd = s4 * 4;
    int h  = hd / D;
    int s = src[e], d = dst[e];
    float alpha = ebuf[e * H + h] / denom[d * H + h];
    float4 fv = ((const float4*)(f + (size_t)s * (H * D)))[s4];
    float* rp = rst + (size_t)d * (H * D) + hd;
    atomicAdd(rp + 0, fv.x * alpha);
    atomicAdd(rp + 1, fv.y * alpha);
    atomicAdd(rp + 2, fv.z * alpha);
    atomicAdd(rp + 3, fv.w * alpha);
}

// ---------------------------------------------------------------------------
// Finalize: out = act(rst + bias [+ res]).  act: 1 = ELU, 0 = identity.
// Safe for out == res (elementwise).
// ---------------------------------------------------------------------------
__global__ void node_finalize(float* __restrict__ out,
                              const float* __restrict__ rst,
                              const float* __restrict__ bias,
                              const float* __restrict__ res,
                              int total, int HD, int act)
{
    int idx = blockIdx.x * blockDim.x + threadIdx.x;
    if (idx >= total) return;
    float v = rst[idx] + bias[idx % HD];
    if (res) v += res[idx];
    if (act) v = v > 0.f ? v : (__expf(v) - 1.f);
    out[idx] = v;
}

// ---------------------------------------------------------------------------
static void gat_attention(const float* f, const float* al, const float* ar,
                          const int* src, const int* dst,
                          float* el, float* er, float* mb, float* dn,
                          float* ebuf, float* rst, int H, int D,
                          hipStream_t s)
{
    const int NH = NN * H, EH = NE * H, NHD = NN * H * D;
    const int aggT = NE * ((H * D) / 4);
    fill_u32<<<(NH + 255) / 256, 256, 0, s>>>((unsigned*)mb, 0xFF800000u, NH);
    fill_u32<<<(NH + 255) / 256, 256, 0, s>>>((unsigned*)dn, 0u, NH);
    fill_u32<<<(NHD + 255) / 256, 256, 0, s>>>((unsigned*)rst, 0u, NHD);
    attn_logits<<<(NH + 255) / 256, 256, 0, s>>>(f, al, ar, el, er, NH, H, D);
    edge_logits_max<<<(EH + 255) / 256, 256, 0, s>>>(ebuf, el, er, src, dst, mb, EH, H);
    edge_exp_sum<<<(EH + 255) / 256, 256, 0, s>>>(ebuf, dst, mb, dn, EH, H);
    edge_aggregate<<<(aggT + 255) / 256, 256, 0, s>>>(rst, f, ebuf, dn, src, dst, H, D, NE);
}

extern "C" void kernel_launch(void* const* d_in, const int* in_sizes, int n_in,
                              void* d_out, int out_size, void* d_ws, size_t ws_size,
                              hipStream_t stream)
{
    const float* x   = (const float*)d_in[0];
    const int*   src = (const int*)d_in[1];
    const int*   dst = (const int*)d_in[2];
    const float* W1  = (const float*)d_in[3];
    const float* al1 = (const float*)d_in[4];
    const float* ar1 = (const float*)d_in[5];
    const float* b1  = (const float*)d_in[6];
    const float* W2  = (const float*)d_in[7];
    const float* al2 = (const float*)d_in[8];
    const float* ar2 = (const float*)d_in[9];
    const float* b2  = (const float*)d_in[10];
    const float* W3  = (const float*)d_in[11];
    const float* al3 = (const float*)d_in[12];
    const float* ar3 = (const float*)d_in[13];
    const float* b3  = (const float*)d_in[14];
    const float* Wr3 = (const float*)d_in[15];
    float* out = (float*)d_out;

    // ---- workspace partition (all offsets 256B aligned) ----
    const size_t NHD = (size_t)NN * 512;         // 25.6M elements
    char* ws = (char*)d_ws;
    float*  fbuf = (float*)ws;                ws += NHD * 4;          // pre-agg features
    float*  hbuf = (float*)ws;                ws += NHD * 4;          // node features h
    float*  rbuf = (float*)ws;                ws += NHD * 4;          // rst accumulator
    __bf16* apk  = (__bf16*)ws;               ws += NHD * 2;          // bf16 A panel
    __bf16* w1p  = (__bf16*)ws;               ws += (size_t)256 * 512 * 2;
    __bf16* w2p  = (__bf16*)ws;               ws += (size_t)512 * 512 * 2;
    __bf16* w3p  = (__bf16*)ws;               ws += (size_t)512 * 64 * 2;
    __bf16* wr3p = (__bf16*)ws;               ws += (size_t)512 * 64 * 2;
    float*  el   = (float*)ws;                ws += (size_t)NN * 8 * 4;
    float*  er   = (float*)ws;                ws += (size_t)NN * 8 * 4;
    float*  mb   = (float*)ws;                ws += (size_t)NN * 8 * 4;
    float*  dn   = (float*)ws;                ws += (size_t)NN * 8 * 4;
    float*  ebuf = (float*)ws;                ws += (size_t)NE * 8 * 4;
    float*  res3 = (float*)ws;                /* NN*64*4 */

    const dim3 blk128(128);
    const int gX = (NN / 16 + 3) / 4;            // 64-row block tiles (tail-guarded)

    // ---- pack all weights into WMMA B-fragment layout (once per call) ----
    pack_W_bf16<<<((256/32)*(512/16)*32 + 255) / 256, 256, 0, stream>>>(W1, w1p, 256, 512);
    pack_W_bf16<<<((512/32)*(512/16)*32 + 255) / 256, 256, 0, stream>>>(W2, w2p, 512, 512);
    pack_W_bf16<<<((512/32)*(64/16)*32 + 255) / 256, 256, 0, stream>>>(W3, w3p, 512, 64);
    pack_W_bf16<<<((512/32)*(64/16)*32 + 255) / 256, 256, 0, stream>>>(Wr3, wr3p, 512, 64);

    // ================= Layer 1: x[50000,256] -> f[50000,8,64] =============
    pack_A_bf16<<<(NN * 256 + 255) / 256, 256, 0, stream>>>(x, apk, NN * 256);
    gemm_wmma_bf16p<<<dim3(gX, 512 / 64), blk128, 0, stream>>>(apk, w1p, fbuf, NN, 512, 256);
    gat_attention(fbuf, al1, ar1, src, dst, el, er, mb, dn, ebuf, rbuf, 8, 64, stream);
    node_finalize<<<((int)NHD + 255) / 256, 256, 0, stream>>>(hbuf, rbuf, b1, nullptr, (int)NHD, 512, 1);

    // ================= Layer 2: h[50000,512] -> f, identity residual ======
    pack_A_bf16<<<((int)NHD + 255) / 256, 256, 0, stream>>>(hbuf, apk, (int)NHD);
    gemm_wmma_bf16p<<<dim3(gX, 512 / 64), blk128, 0, stream>>>(apk, w2p, fbuf, NN, 512, 512);
    gat_attention(fbuf, al2, ar2, src, dst, el, er, mb, dn, ebuf, rbuf, 8, 64, stream);
    node_finalize<<<((int)NHD + 255) / 256, 256, 0, stream>>>(hbuf, rbuf, b2, hbuf, (int)NHD, 512, 1);

    // ================= Layer 3: h[50000,512] -> f[50000,1,64] =============
    pack_A_bf16<<<((int)NHD + 255) / 256, 256, 0, stream>>>(hbuf, apk, (int)NHD);
    gemm_wmma_bf16p<<<dim3(gX, 1), blk128, 0, stream>>>(apk, wr3p, res3, NN, 64, 512); // proj residual
    gemm_wmma_bf16p<<<dim3(gX, 1), blk128, 0, stream>>>(apk, w3p, fbuf, NN, 64, 512);
    gat_attention(fbuf, al3, ar3, src, dst, el, er, mb, dn, ebuf, rbuf, 1, 64, stream);
    // 1 head: mean over heads == the head itself; no activation.
    node_finalize<<<(NN * 64 + 255) / 256, 256, 0, stream>>>(out, rbuf, b3, res3, NN * 64, 64, 0);
}
